// AGGate_88287347737087
// MI455X (gfx1250) — matrix operand order repaired and checked by reading, not compiled, not run on hardware
//
#include <hip/hip_runtime.h>
#include <hip/hip_bf16.h>
#include <math.h>

typedef __attribute__((ext_vector_type(16))) _Float16 v16h;
typedef __attribute__((ext_vector_type(8)))  _Float16 v8h;
typedef __attribute__((ext_vector_type(8)))  float    v8f;

#define B_  8
#define L_  8192
#define D_  512
#define M_TOTAL (B_ * L_)          // 65536 rows
#define ROWS_PER_WG 64
#define A_STRIDE 520               // 512 + 8 halves pad -> conflict-free ds_load_b128

// ---------------------------------------------------------------------------
// Kernel 1: eqb[b][n] = emotion[b,:] @ W1[:,n] + b1[n] + b2[n]
// (b3 dropped: softmax is invariant to a constant logit shift)
// ---------------------------------------------------------------------------
__global__ void eqb_kernel(const float* __restrict__ emo,
                           const float* __restrict__ W1,
                           const float* __restrict__ b1,
                           const float* __restrict__ b2,
                           float* __restrict__ eqb) {
    int n = blockIdx.x * blockDim.x + threadIdx.x;   // 0..511
    int b = blockIdx.y;                              // 0..7
    const float* e = emo + b * D_;
    float acc = 0.f;
    #pragma unroll 8
    for (int k = 0; k < D_; ++k)
        acc += e[k] * W1[(size_t)k * D_ + n];
    eqb[b * D_ + n] = acc + b1[n] + b2[n];
}

// ---------------------------------------------------------------------------
// Kernel 2: repack W2 (f32, KxN row-major) into f16 WMMA B-fragments.
// Fragment (kt,nt): 32 lanes x 16 halves contiguous.
//   lane l, half h -> W2[K = kt*32 + (l>>4)*16 + h, N = nt*16 + (l&15)]
// ---------------------------------------------------------------------------
__global__ void repack_w2(const float* __restrict__ W2, _Float16* __restrict__ w2p) {
    int idx = blockIdx.x * blockDim.x + threadIdx.x;  // 512 frags * 32 lanes
    if (idx >= 512 * 32) return;
    int frag = idx >> 5, lane = idx & 31;
    int kt = frag >> 5, nt = frag & 31;
    int kbase = kt * 32 + ((lane >> 4) * 16);
    int n = nt * 16 + (lane & 15);
    _Float16* dst = w2p + (size_t)idx * 16;
    #pragma unroll
    for (int h = 0; h < 16; ++h)
        dst[h] = (_Float16)W2[(size_t)(kbase + h) * D_ + n];
}

// ---------------------------------------------------------------------------
// Kernel 3: fused logits GEMM.
// logits[row] = sum_n tanh( (sents[row,:] @ W2)[n] + eqb[b,n] ) * W3[n]
// 1024 WGs x 128 threads (4 waves); each wave owns a 16-row M-tile.
// ---------------------------------------------------------------------------
__global__ __launch_bounds__(128) void attn_logits(
    const float*    __restrict__ sents,
    const _Float16* __restrict__ w2p,
    const float*    __restrict__ eqb,
    const float*    __restrict__ W3,
    float*          __restrict__ logits) {

    __shared__ __align__(32) _Float16 a_lds[ROWS_PER_WG * A_STRIDE]; // 66560 B
    __shared__ __align__(32) _Float16 b_lds[16 * 512];               // 16384 B
    __shared__ float lpart[ROWS_PER_WG];

    const int tid = threadIdx.x;
    const int rowBase = blockIdx.x * ROWS_PER_WG;
    const int b = rowBase >> 13;                 // / 8192 (64 | 8192)

    // Stage A: 64 rows x 512 f32 -> f16 into padded LDS (row-major).
    for (int i = tid; i < ROWS_PER_WG * 128; i += 128) {
        int row = i >> 7;
        int c4  = (i & 127) * 4;
        const float4 v = *(const float4*)(sents + (size_t)(rowBase + row) * D_ + c4);
        _Float16* d = &a_lds[row * A_STRIDE + c4];
        d[0] = (_Float16)v.x; d[1] = (_Float16)v.y;
        d[2] = (_Float16)v.z; d[3] = (_Float16)v.w;
    }
    if (tid < ROWS_PER_WG) lpart[tid] = 0.f;

    const int wave = tid >> 5, lane = tid & 31;
    const int l15 = lane & 15, grp = lane >> 4;
    const _Float16* abase = &a_lds[(wave * 16 + l15) * A_STRIDE + grp * 16];

    for (int nt = 0; nt < 32; ++nt) {
        // Stage B block (all kt for this nt): 16 frags * 1KB = 16KB, shared by 4 waves.
        for (int ch = tid; ch < 512; ch += 128) {
            int kt = ch >> 5, cc = ch & 31;
            v16h val = *(const v16h*)(w2p + ((size_t)(kt * 32 + nt) * 32 + cc) * 16);
            *(v16h*)(&b_lds[kt * 512 + cc * 16]) = val;
        }
        __syncthreads();   // covers A staging (first iter) + B block ready

        const float eqv = eqb[b * D_ + nt * 16 + l15];
        const float w3w = W3[nt * 16 + l15];
        v8f acc = {};
        #pragma unroll
        for (int kt = 0; kt < 16; ++kt) {
            // A fragment (16x32 f16): lane l, halves 0..15 -> K=kb..kb+15,
            // kb = kt*32 + (l>>4)*16.
            v8h a0 = *(const v8h*)(abase + kt * 32);
            v8h a1 = *(const v8h*)(abase + kt * 32 + 8);
            v16h a = __builtin_shufflevector(a0, a1, 0,1,2,3,4,5,6,7,
                                                     8,9,10,11,12,13,14,15);
            v16h bf = *(const v16h*)(&b_lds[kt * 512 + lane * 16]);
            acc = __builtin_amdgcn_wmma_f32_16x16x32_f16(
                false, a, false, bf, (short)0, acc, false, false);
        }
        // acc[r] holds C[M = r + 8*grp, N = nt*16 + l15]; reduce over N.
        #pragma unroll
        for (int r = 0; r < 8; ++r) {
            float t = tanhf(acc[r] + eqv) * w3w;
            atomicAdd(&lpart[wave * 16 + r + grp * 8], t);  // ds_add_f32
        }
        __syncthreads();   // protect b_lds before next nt overwrites it
    }

    if (tid < ROWS_PER_WG) logits[rowBase + tid] = lpart[tid];
}

// ---------------------------------------------------------------------------
// Kernel 4: per-batch softmax + inclusive cumsum over L=8192.
// One 1024-thread block per batch; 8 elements per thread.
// ---------------------------------------------------------------------------
__global__ __launch_bounds__(1024) void softmax_cumsum(
    const float* __restrict__ logits,
    float* __restrict__ s_out,
    float* __restrict__ c_out) {

    __shared__ float red[32];
    __shared__ float wsum[32];

    const int b = blockIdx.x, tid = threadIdx.x;
    const int warp = tid >> 5, lane = tid & 31;
    const float* x = logits + (size_t)b * L_ + tid * 8;

    float v[8];
    float m = -3.402823e38f;
    #pragma unroll
    for (int i = 0; i < 8; ++i) { v[i] = x[i]; m = fmaxf(m, v[i]); }

    // Block max
    #pragma unroll
    for (int off = 16; off > 0; off >>= 1) m = fmaxf(m, __shfl_xor(m, off, 32));
    if (lane == 0) red[warp] = m;
    __syncthreads();
    if (tid < 32) {
        float t = red[tid];
        #pragma unroll
        for (int off = 16; off > 0; off >>= 1) t = fmaxf(t, __shfl_xor(t, off, 32));
        if (tid == 0) red[0] = t;
    }
    __syncthreads();
    m = red[0];

    // exp + thread-local inclusive prefix
    float e[8], run = 0.f;
    #pragma unroll
    for (int i = 0; i < 8; ++i) { e[i] = __expf(v[i] - m); run += e[i]; v[i] = run; }
    const float tot = run;

    // Warp inclusive scan of thread totals
    float sc = tot;
    #pragma unroll
    for (int off = 1; off < 32; off <<= 1) {
        float t = __shfl_up(sc, off, 32);
        if (lane >= off) sc += t;
    }
    if (lane == 31) wsum[warp] = sc;
    __syncthreads();
    if (tid < 32) {
        float t = wsum[tid];
        #pragma unroll
        for (int off = 1; off < 32; off <<= 1) {
            float u = __shfl_up(t, off, 32);
            if (tid >= off) t += u;
        }
        wsum[tid] = t;
    }
    __syncthreads();

    const float base  = (warp ? wsum[warp - 1] : 0.f) + (sc - tot);
    const float total = wsum[31];
    const float inv   = 1.f / total;

    float* so = s_out + (size_t)b * L_ + tid * 8;
    float* co = c_out + (size_t)b * L_ + tid * 8;
    #pragma unroll
    for (int i = 0; i < 8; ++i) {
        so[i] = e[i] * inv;               // softmax s
        co[i] = (base + v[i]) * inv;      // inclusive cumsum c
    }
}

// ---------------------------------------------------------------------------
// Kernel 5: out = (c-s)*pre + s*sents + (1-c)*pos   (float4 vectorized)
// ---------------------------------------------------------------------------
__global__ void combine(const float* __restrict__ s,
                        const float* __restrict__ c,
                        const float* __restrict__ pre,
                        const float* __restrict__ sents,
                        const float* __restrict__ pos,
                        float* __restrict__ out) {
    const long n4 = (long)M_TOTAL * (D_ / 4);        // 8388608 float4s
    const long stride = (long)gridDim.x * blockDim.x;
    for (long i = (long)blockIdx.x * blockDim.x + threadIdx.x; i < n4; i += stride) {
        long j = i >> 7;                              // 128 float4s per (b,l)
        float sv = s[j], cv = c[j];
        float pw = cv - sv, ow = 1.f - cv;
        float4 a = ((const float4*)pre)[i];
        float4 q = ((const float4*)sents)[i];
        float4 d = ((const float4*)pos)[i];
        float4 r;
        r.x = pw * a.x + sv * q.x + ow * d.x;
        r.y = pw * a.y + sv * q.y + ow * d.y;
        r.z = pw * a.z + sv * q.z + ow * d.z;
        r.w = pw * a.w + sv * q.w + ow * d.w;
        ((float4*)out)[i] = r;
    }
}

// ---------------------------------------------------------------------------
extern "C" void kernel_launch(void* const* d_in, const int* in_sizes, int n_in,
                              void* d_out, int out_size, void* d_ws, size_t ws_size,
                              hipStream_t stream) {
    const float* emotion = (const float*)d_in[0];
    const float* sents   = (const float*)d_in[1];
    const float* pre     = (const float*)d_in[2];
    const float* pos     = (const float*)d_in[3];
    const float* W1      = (const float*)d_in[4];
    const float* b1      = (const float*)d_in[5];
    const float* W2      = (const float*)d_in[6];
    const float* b2      = (const float*)d_in[7];
    const float* W3      = (const float*)d_in[8];
    // d_in[9] = b3: constant logit shift, softmax-invariant -> unused.

    char* ws = (char*)d_ws;
    _Float16* w2p    = (_Float16*)(ws);                 // 512*512*2 = 524288 B
    float*    eqb    = (float*)(ws + 524288);           // 8*512*4   =  16384 B
    float*    logits = (float*)(ws + 540672);           // 65536*4   = 262144 B
    float*    sArr   = (float*)(ws + 802816);           // 262144 B
    float*    cArr   = (float*)(ws + 1064960);          // 262144 B  (total ~1.27 MB)

    eqb_kernel   <<<dim3(2, 8), 256, 0, stream>>>(emotion, W1, b1, b2, eqb);
    repack_w2    <<<64, 256, 0, stream>>>(W2, w2p);
    attn_logits  <<<M_TOTAL / ROWS_PER_WG, 128, 0, stream>>>(sents, w2p, eqb, W3, logits);
    softmax_cumsum<<<B_, 1024, 0, stream>>>(logits, sArr, cArr);
    combine      <<<4096, 256, 0, stream>>>(sArr, cArr, pre, sents, pos, (float*)d_out);
}
